// GraphormerLinearAttention_15779709846382
// MI455X (gfx1250) — matrix operand order repaired and checked by reading, not compiled, not run on hardware
//
#include <hip/hip_runtime.h>
#include <hip/hip_bf16.h>

typedef _Float16 h4  __attribute__((ext_vector_type(4)));
typedef _Float16 h8  __attribute__((ext_vector_type(8)));
typedef _Float16 v16h __attribute__((ext_vector_type(16)));
typedef float    v8f  __attribute__((ext_vector_type(8)));

#define B_   8
#define N_   2048
#define H_   512
#define NH_  16
#define HD_  32
#define ROWS (B_ * N_)        /* 16384 */
#define EPS_ 1e-9f

// ---------------- WMMA fragment helpers (gfx1250 wave32 layouts) ----------------
// A (16x32 f16): lane L -> row r=L&15; halves[0..7]=A[r,k0..k0+7], halves[8..15]=A[r,k0+16..k0+23], k0=(L>=16)?8:0
__device__ __forceinline__ v16h frag_a(const _Float16* base, int ld, int lane) {
    int r  = lane & 15;
    int k0 = (lane & 16) ? 8 : 0;
    h8 x0 = *(const h8*)(base + r * ld + k0);
    h8 x1 = *(const h8*)(base + r * ld + k0 + 16);
    v16h a;
#pragma unroll
    for (int i = 0; i < 8; ++i) { a[i] = x0[i]; a[8 + i] = x1[i]; }
    return a;
}
// B (32x16 f16), supplied TRANSPOSED in memory: baseT[n][k].
// lane L -> col n=L&15; halves[0..7]=B[k0..k0+7,n], halves[8..15]=B[k0+16..k0+23,n], k0=(L>=16)?8:0
__device__ __forceinline__ v16h frag_b(const _Float16* baseT, int ld, int lane) {
    int n  = lane & 15;
    int k0 = (lane & 16) ? 8 : 0;
    h8 x0 = *(const h8*)(baseT + n * ld + k0);
    h8 x1 = *(const h8*)(baseT + n * ld + k0 + 16);
    v16h b;
#pragma unroll
    for (int i = 0; i < 8; ++i) { b[i] = x0[i]; b[8 + i] = x1[i]; }
    return b;
}
__device__ __forceinline__ v8f wmma_f16(v16h a, v16h b, v8f c) {
    return __builtin_amdgcn_wmma_f32_16x16x32_f16(false, a, false, b, (short)0, c, false, false);
}
__device__ __forceinline__ float elu1(float x) { return x > 0.f ? x + 1.f : __expf(x); }

// ---------------- K0: f32 -> f16 convert ----------------
__global__ void cvt_k(_Float16* __restrict__ dst, const float* __restrict__ src, int n) {
    int i = (blockIdx.x * 256 + threadIdx.x) * 4;
    if (i < n) {
        float4 v = *(const float4*)(src + i);
        h4 o; o[0] = (_Float16)v.x; o[1] = (_Float16)v.y; o[2] = (_Float16)v.z; o[3] = (_Float16)v.w;
        *(h4*)(dst + i) = o;
    }
}

// ---------------- K1: bias = mean(encoding_bias, axis=-1) ----------------
__global__ void bias_mean_k(const float* __restrict__ eb, float* __restrict__ bias) {
    int bn = blockIdx.x;                       // 0..16383
    const float* row = eb + (size_t)bn * N_;
    int t = threadIdx.x;
    float s = 0.f;
#pragma unroll
    for (int j = 0; j < N_ / 256; ++j) s += row[t + j * 256];
#pragma unroll
    for (int off = 16; off; off >>= 1) s += __shfl_xor(s, off);
    __shared__ float ps[8];
    if ((t & 31) == 0) ps[t >> 5] = s;
    __syncthreads();
    if (t == 0) {
        float tot = 0.f;
#pragma unroll
        for (int w = 0; w < 8; ++w) tot += ps[w];
        bias[bn] = tot * (1.0f / N_);
    }
}

// ---------------- K2: fused QKV projections + feature map (double-buffered LDS) ----------------
// out = elu(x @ W^T [+ bias | + bv]) + 1, stored f16. blockIdx.z selects Q/K/V.
__global__ void __launch_bounds__(256) qkv_k(
    const _Float16* __restrict__ xh,
    const _Float16* __restrict__ Wqh, const _Float16* __restrict__ Wkh, const _Float16* __restrict__ Wvh,
    const float* __restrict__ bias, const float* __restrict__ bv,
    _Float16* __restrict__ qh, _Float16* __restrict__ kh, _Float16* __restrict__ vh)
{
    __shared__ _Float16 As[2][64 * 32];     // 64 rows  x K32
    __shared__ _Float16 Bs[2][128 * 32];    // 128 cols x K32 (row-major W == transposed B)
    int t = threadIdx.x, lane = t & 31, w = t >> 5;
    int m0 = blockIdx.x * 64, n0 = blockIdx.y * 128;
    int which = blockIdx.z;
    const _Float16* W = (which == 0) ? Wqh : (which == 1) ? Wkh : Wvh;
    _Float16* out     = (which == 0) ? qh  : (which == 1) ? kh  : vh;

    int row = t >> 2, seg = t & 3;              // A-tile copy coords
    int brow0 = t >> 2,          bseg0 = t & 3; // B-tile copy coords (2 slabs)
    int brow1 = (t + 256) >> 2,  bseg1 = t & 3;
    const _Float16* aptr  = xh + (size_t)(m0 + row)  * H_ + seg * 8;
    const _Float16* bptr0 = W  + (size_t)(n0 + brow0) * H_ + bseg0 * 8;
    const _Float16* bptr1 = W  + (size_t)(n0 + brow1) * H_ + bseg1 * 8;

    int mt = w & 3, nh = w >> 2;                // wave tile: 16 rows x 64 cols
    v8f acc[4] = {};

    // preload k-chunk 0 into buffer 0
    {
        h8 ax  = *(const h8*)(aptr);
        h8 bx0 = *(const h8*)(bptr0);
        h8 bx1 = *(const h8*)(bptr1);
        *(h8*)(&As[0][row * 32 + seg * 8])    = ax;
        *(h8*)(&Bs[0][brow0 * 32 + bseg0 * 8]) = bx0;
        *(h8*)(&Bs[0][brow1 * 32 + bseg1 * 8]) = bx1;
    }
    __syncthreads();

    for (int kb = 0; kb < H_; kb += 32) {
        int  cur  = (kb >> 5) & 1;
        bool more = (kb + 32) < H_;
        h8 ax = {}, bx0 = {}, bx1 = {};
        if (more) {                              // issue next tile's global loads early
            ax  = *(const h8*)(aptr  + kb + 32);
            bx0 = *(const h8*)(bptr0 + kb + 32);
            bx1 = *(const h8*)(bptr1 + kb + 32);
        }
        v16h a = frag_a(&As[cur][mt * 16 * 32], 32, lane);
#pragma unroll
        for (int j = 0; j < 4; ++j) {
            v16h b = frag_b(&Bs[cur][(nh * 64 + j * 16) * 32], 32, lane);
            acc[j] = wmma_f16(a, b, acc[j]);
        }
        if (more) {                              // stage into the idle buffer
            *(h8*)(&As[cur ^ 1][row * 32 + seg * 8])     = ax;
            *(h8*)(&Bs[cur ^ 1][brow0 * 32 + bseg0 * 8]) = bx0;
            *(h8*)(&Bs[cur ^ 1][brow1 * 32 + bseg1 * 8]) = bx1;
        }
        __syncthreads();
    }

    int rbase = m0 + mt * 16 + (lane >> 4) * 8;
    int cbase = n0 + nh * 64 + (lane & 15);
#pragma unroll
    for (int j = 0; j < 4; ++j) {
        int col = cbase + j * 16;
        float addc = (which == 2) ? bv[col] : 0.f;
#pragma unroll
        for (int i = 0; i < 8; ++i) {
            int r = rbase + i;
            float v = acc[j][i] + addc + ((which == 0) ? bias[r] : 0.f);
            out[(size_t)r * H_ + col] = (_Float16)elu1(v);
        }
    }
}

// ---------------- K3: kv = V^T @ K per (b,h) + ksum (double-buffered LDS) ----------------
__global__ void __launch_bounds__(128) kv_k(
    const _Float16* __restrict__ kh, const _Float16* __restrict__ vh,
    _Float16* __restrict__ kvh, float* __restrict__ ksum)
{
    __shared__ _Float16 Kt[2][32 * 40];   // Kt[buf][d][n] (transposed on store)
    __shared__ _Float16 Vt[2][32 * 40];   // Vt[buf][m][n]
    __shared__ float ks[32];
    int t = threadIdx.x, lane = t & 31, w = t >> 5;
    int bh = blockIdx.x;               // 0..127
    int b = bh >> 4, h = bh & 15;
    if (t < 32) ks[t] = 0.f;

    int nrow = t >> 2, seg = t & 3;    // cooperative-load coords
    size_t base = (size_t)b * N_ * H_ + (size_t)h * HD_;
    float ksl[8];
#pragma unroll
    for (int j = 0; j < 8; ++j) ksl[j] = 0.f;
    v8f acc = {};
    int mt = w >> 1, nt = w & 1;       // 2x2 WMMA tiles over (32 x 32) output

    // preload chunk 0 into buffer 0 (transpose-on-store, accumulate ksum)
    {
        size_t roff = base + (size_t)nrow * H_ + seg * 8;
        h8 kx = *(const h8*)(kh + roff);
        h8 vx = *(const h8*)(vh + roff);
#pragma unroll
        for (int j = 0; j < 8; ++j) {
            int d = seg * 8 + j;
            Kt[0][d * 40 + nrow] = kx[j];
            Vt[0][d * 40 + nrow] = vx[j];
            ksl[j] += (float)kx[j];
        }
    }
    __syncthreads();

    for (int nb = 0; nb < N_; nb += 32) {
        int  cur  = (nb >> 5) & 1;
        bool more = (nb + 32) < N_;
        h8 kx = {}, vx = {};
        if (more) {                                        // fetch next chunk early
            size_t roff = base + (size_t)(nb + 32 + nrow) * H_ + seg * 8;
            kx = *(const h8*)(kh + roff);
            vx = *(const h8*)(vh + roff);
        }
        v16h a  = frag_a(&Vt[cur][mt * 16 * 40], 40, lane);   // A[m,kk] = v[nb+kk, m]
        v16h bf = frag_b(&Kt[cur][nt * 16 * 40], 40, lane);   // B[kk,d] = k[nb+kk, d]
        acc = wmma_f16(a, bf, acc);
        if (more) {                                        // transpose-store into idle buffer
#pragma unroll
            for (int j = 0; j < 8; ++j) {
                int d = seg * 8 + j;
                Kt[cur ^ 1][d * 40 + nrow] = kx[j];
                Vt[cur ^ 1][d * 40 + nrow] = vx[j];
                ksl[j] += (float)kx[j];
            }
        }
        __syncthreads();
    }
#pragma unroll
    for (int j = 0; j < 8; ++j) atomicAdd(&ks[seg * 8 + j], ksl[j]);
    __syncthreads();
    if (t < 32) ksum[bh * 32 + t] = ks[t];

    int m = mt * 16 + (lane >> 4) * 8;
    int d = nt * 16 + (lane & 15);
#pragma unroll
    for (int i = 0; i < 8; ++i)
        kvh[(size_t)(bh * 32 + m + i) * 32 + d] = (_Float16)acc[i];   // row-major (m,d) == Bt for attn
}

// ---------------- K4: attn = (q @ kv^T) * z, masked ----------------
__global__ void __launch_bounds__(128) attn_k(
    const _Float16* __restrict__ qh, const _Float16* __restrict__ kvh,
    const float* __restrict__ ksum, _Float16* __restrict__ attnh)
{
    __shared__ float ksS[32];
    __shared__ float zS[64];
    int t = threadIdx.x, lane = t & 31, w = t >> 5;
    int n0 = blockIdx.x * 64;
    int bh = blockIdx.y;
    int b = bh >> 4, h = bh & 15;
    if (t < 32) ksS[t] = ksum[bh * 32 + t];
    __syncthreads();
    {   // per-row z = 1/(q . ksum + eps); padding mask via max(q)==0
        int r = t >> 1, hf = t & 1;
        size_t qoff = (size_t)(b * N_ + n0 + r) * H_ + h * HD_ + hf * 16;
        h8 q0 = *(const h8*)(qh + qoff);
        h8 q1 = *(const h8*)(qh + qoff + 8);
        float p = 0.f, mx = 0.f;
#pragma unroll
        for (int j = 0; j < 8; ++j) {
            float a0 = (float)q0[j], a1 = (float)q1[j];
            p += a0 * ksS[hf * 16 + j] + a1 * ksS[hf * 16 + 8 + j];
            mx = fmaxf(mx, fmaxf(a0, a1));
        }
        p  += __shfl_xor(p, 1);
        mx  = fmaxf(mx, __shfl_xor(mx, 1));
        if (hf == 0) zS[r] = (mx > 0.f) ? (1.f / (p + EPS_)) : 0.f;
    }
    __syncthreads();

    const _Float16* qbase = qh + (size_t)(b * N_ + n0 + w * 16) * H_ + h * HD_;
    v16h a = frag_a(qbase, H_, lane);                    // K = full head dim (32) in one WMMA
#pragma unroll
    for (int nt = 0; nt < 2; ++nt) {
        v16h bf = frag_b(kvh + (size_t)bh * 1024 + nt * 16 * 32, 32, lane);
        v8f c = {};
        c = wmma_f16(a, bf, c);
        int rl  = w * 16 + (lane >> 4) * 8;
        int col = h * HD_ + nt * 16 + (lane & 15);
#pragma unroll
        for (int i = 0; i < 8; ++i) {
            float vv = c[i] * zS[rl + i];
            attnh[(size_t)(b * N_ + n0 + rl + i) * H_ + col] = (_Float16)vv;
        }
    }
}

// ---------------- K5: out = attn @ Wo^T + bo (f32 out, double-buffered LDS) ----------------
__global__ void __launch_bounds__(256) out_k(
    const _Float16* __restrict__ attnh, const _Float16* __restrict__ Woh,
    const float* __restrict__ bo, float* __restrict__ out)
{
    __shared__ _Float16 As[2][64 * 32];
    __shared__ _Float16 Bs[2][128 * 32];
    int t = threadIdx.x, lane = t & 31, w = t >> 5;
    int m0 = blockIdx.x * 64, n0 = blockIdx.y * 128;
    int mt = w & 3, nh = w >> 2;

    int row = t >> 2, seg = t & 3;
    int brow0 = t >> 2,          bseg0 = t & 3;
    int brow1 = (t + 256) >> 2,  bseg1 = t & 3;
    const _Float16* aptr  = attnh + (size_t)(m0 + row)  * H_ + seg * 8;
    const _Float16* bptr0 = Woh   + (size_t)(n0 + brow0) * H_ + bseg0 * 8;
    const _Float16* bptr1 = Woh   + (size_t)(n0 + brow1) * H_ + bseg1 * 8;

    v8f acc[4] = {};
    {
        h8 ax  = *(const h8*)(aptr);
        h8 bx0 = *(const h8*)(bptr0);
        h8 bx1 = *(const h8*)(bptr1);
        *(h8*)(&As[0][row * 32 + seg * 8])     = ax;
        *(h8*)(&Bs[0][brow0 * 32 + bseg0 * 8]) = bx0;
        *(h8*)(&Bs[0][brow1 * 32 + bseg1 * 8]) = bx1;
    }
    __syncthreads();

    for (int kb = 0; kb < H_; kb += 32) {
        int  cur  = (kb >> 5) & 1;
        bool more = (kb + 32) < H_;
        h8 ax = {}, bx0 = {}, bx1 = {};
        if (more) {
            ax  = *(const h8*)(aptr  + kb + 32);
            bx0 = *(const h8*)(bptr0 + kb + 32);
            bx1 = *(const h8*)(bptr1 + kb + 32);
        }
        v16h a = frag_a(&As[cur][mt * 16 * 32], 32, lane);
#pragma unroll
        for (int j = 0; j < 4; ++j) {
            v16h b = frag_b(&Bs[cur][(nh * 64 + j * 16) * 32], 32, lane);
            acc[j] = wmma_f16(a, b, acc[j]);
        }
        if (more) {
            *(h8*)(&As[cur ^ 1][row * 32 + seg * 8])     = ax;
            *(h8*)(&Bs[cur ^ 1][brow0 * 32 + bseg0 * 8]) = bx0;
            *(h8*)(&Bs[cur ^ 1][brow1 * 32 + bseg1 * 8]) = bx1;
        }
        __syncthreads();
    }

    int rbase = m0 + mt * 16 + (lane >> 4) * 8;
    int cb    = n0 + nh * 64 + (lane & 15);
#pragma unroll
    for (int j = 0; j < 4; ++j) {
        int col = cb + j * 16;
        float bb = bo[col];
#pragma unroll
        for (int i = 0; i < 8; ++i)
            out[(size_t)(rbase + i) * H_ + col] = acc[j][i] + bb;
    }
}

extern "C" void kernel_launch(void* const* d_in, const int* in_sizes, int n_in,
                              void* d_out, int out_size, void* d_ws, size_t ws_size,
                              hipStream_t stream)
{
    (void)in_sizes; (void)n_in; (void)out_size; (void)ws_size;
    const float* x  = (const float*)d_in[0];
    const float* eb = (const float*)d_in[1];
    const float* Wq = (const float*)d_in[2];
    const float* Wk = (const float*)d_in[3];
    const float* Wv = (const float*)d_in[4];
    const float* bv = (const float*)d_in[5];
    const float* Wo = (const float*)d_in[6];
    const float* bo = (const float*)d_in[7];
    float* out = (float*)d_out;

    char* ws = (char*)d_ws;
    size_t off = 0;
    auto alloc = [&](size_t bytes) { void* p = ws + off; off += (bytes + 255) & ~(size_t)255; return p; };
    float*    bias = (float*)   alloc((size_t)ROWS * 4);
    _Float16* xh   = (_Float16*)alloc((size_t)ROWS * H_ * 2);
    _Float16* Wqh  = (_Float16*)alloc((size_t)H_ * H_ * 2);
    _Float16* Wkh  = (_Float16*)alloc((size_t)H_ * H_ * 2);
    _Float16* Wvh  = (_Float16*)alloc((size_t)H_ * H_ * 2);
    _Float16* Woh  = (_Float16*)alloc((size_t)H_ * H_ * 2);
    _Float16* qh   = (_Float16*)alloc((size_t)ROWS * H_ * 2);
    _Float16* kh   = (_Float16*)alloc((size_t)ROWS * H_ * 2);
    _Float16* vh   = (_Float16*)alloc((size_t)ROWS * H_ * 2);
    _Float16* kvh  = (_Float16*)alloc((size_t)128 * 32 * 32 * 2);
    float*    ksum = (float*)   alloc((size_t)128 * 32 * 4);
    _Float16* attnh = xh;   // xh is dead after qkv_k; reuse for attn intermediate

    cvt_k<<<(ROWS * H_) / 1024, 256, 0, stream>>>(xh,  x,  ROWS * H_);
    cvt_k<<<(H_ * H_) / 1024,  256, 0, stream>>>(Wqh, Wq, H_ * H_);
    cvt_k<<<(H_ * H_) / 1024,  256, 0, stream>>>(Wkh, Wk, H_ * H_);
    cvt_k<<<(H_ * H_) / 1024,  256, 0, stream>>>(Wvh, Wv, H_ * H_);
    cvt_k<<<(H_ * H_) / 1024,  256, 0, stream>>>(Woh, Wo, H_ * H_);
    bias_mean_k<<<ROWS, 256, 0, stream>>>(eb, bias);
    qkv_k<<<dim3(ROWS / 64, H_ / 128, 3), 256, 0, stream>>>(xh, Wqh, Wkh, Wvh, bias, bv, qh, kh, vh);
    kv_k<<<B_ * NH_, 128, 0, stream>>>(kh, vh, kvh, ksum);
    attn_k<<<dim3(N_ / 64, B_ * NH_), 128, 0, stream>>>(qh, kvh, ksum, attnh);
    out_k<<<dim3(ROWS / 64, H_ / 128), 256, 0, stream>>>(attnh, Woh, bo, out);
}
